// AdvancedCrossStreamAttention_14912126452100
// MI455X (gfx1250) — matrix operand order repaired and checked by reading, not compiled, not run on hardware
//
#include <hip/hip_runtime.h>
#include <hip/hip_bf16.h>

typedef __attribute__((ext_vector_type(16))) _Float16 v16h;
typedef __attribute__((ext_vector_type(8)))  _Float16 v8h;
typedef __attribute__((ext_vector_type(8)))  float    v8f;

#define B_   16
#define C_   256
#define C8_  32
#define N_   1600

union V16 { v16h v; v8h h[2]; _Float16 e[16]; };

__device__ __forceinline__ v8f wmma_f16(v16h a, v16h b, v8f c) {
    // D = A(16x32 f16) x B(32x16 f16) + C(16x16 f32)
    return __builtin_amdgcn_wmma_f32_16x16x32_f16(
        /*neg_a=*/false, a, /*neg_b=*/false, b,
        /*c_mod=*/(short)0, c, /*reuse_a=*/false, /*reuse_b=*/false);
}

// Async copy of one 64-byte row (32 f16) from global memory directly into LDS.
// INST_OFFSET is added to BOTH the LDS and global addresses (ISA 15.18.3),
// so a single (dst,src) pair covers all four 16B chunks. Tracked by ASYNCcnt.
__device__ __forceinline__ void async_row64B(_Float16* dst_lds, const _Float16* src_glb) {
    uint32_t d = (uint32_t)(uintptr_t)dst_lds;   // flat-shared low 32 bits = LDS offset
    asm volatile("global_load_async_to_lds_b128 %0, %1, off"           :: "v"(d), "v"(src_glb) : "memory");
    asm volatile("global_load_async_to_lds_b128 %0, %1, off offset:16" :: "v"(d), "v"(src_glb) : "memory");
    asm volatile("global_load_async_to_lds_b128 %0, %1, off offset:32" :: "v"(d), "v"(src_glb) : "memory");
    asm volatile("global_load_async_to_lds_b128 %0, %1, off offset:48" :: "v"(d), "v"(src_glb) : "memory");
}

// ---------------------------------------------------------------------------
// Projection GEMM: out[o,n] = sum_c W[o,c] * X[b,c,n] + bias[o], cast to f16.
// X tile (32c x 64n) is staged through LDS with coalesced float4 loads and a
// f16 transpose-store so each lane's WMMA B-operand is two ds_load_b128.
// Grid: (N/64, M/16, 3*B). Block: 128 (4 waves; each wave one 16x16 tile).
// ---------------------------------------------------------------------------
__global__ __launch_bounds__(128)
void proj_kernel(const float* __restrict__ s0, const float* __restrict__ s1,
                 const float* __restrict__ s2,
                 const float* __restrict__ W, const float* __restrict__ bias,
                 _Float16* __restrict__ outbase,
                 int M, int shift, int transpose_out)
{
    __shared__ _Float16 xt[64][40];     // [n][c] tile, padded rows

    const int lane = threadIdx.x & 31;
    const int wv   = threadIdx.x >> 5;
    const int lr   = lane & 15;   // row/col within tile
    const int lg   = lane >> 4;   // lane group (0/1)

    const int z = blockIdx.z;           // z = i*B + b
    const int i = z / B_, b = z % B_;
    const int src = (i + shift) % 3;
    const float* X = (src == 0 ? s0 : (src == 1 ? s1 : s2)) + (size_t)b * C_ * N_;

    const int nblk  = blockIdx.x * 64;
    const int nbase = nblk + wv * 16;
    const int obase = blockIdx.y * 16;

    v8f acc = (v8f){0.f,0.f,0.f,0.f,0.f,0.f,0.f,0.f};

    for (int kt = 0; kt < C_ / 32; ++kt) {
        const int cbase = kt * 32;

        __syncthreads();                // close previous iteration's reads
        // Stage X[cbase:+32, nblk:+64] -> xt[n][c]; 4 float4 chunks / thread
        #pragma unroll
        for (int u = 0; u < 4; ++u) {
            const int chunk = threadIdx.x * 4 + u;   // 512 chunks total
            const int c = chunk >> 4;                // 0..31
            const int q = chunk & 15;                // n quad 0..15
            const float4 f = *(const float4*)(X + (size_t)(cbase + c) * N_ + nblk + q * 4);
            xt[q * 4 + 0][c] = (_Float16)f.x;
            xt[q * 4 + 1][c] = (_Float16)f.y;
            xt[q * 4 + 2][c] = (_Float16)f.z;
            xt[q * 4 + 3][c] = (_Float16)f.w;
        }
        __syncthreads();

        V16 a, bb;
        // A operand (16x32, o x c): lane holds row o=obase+lr; two contiguous
        // 8-float chunks of W (compiler emits b128 pairs).
        const float* wrow = W + (size_t)(obase + lr) * C_ + cbase + lg * 8;
        #pragma unroll
        for (int t = 0; t < 8; ++t) {
            a.e[t]     = (_Float16)wrow[t];
            a.e[t + 8] = (_Float16)wrow[16 + t];
        }
        // B operand (32x16, c x n): lane col n, two 8-half c chunks (8lg, 16+8lg) in LDS
        const int nloc = wv * 16 + lr;
        bb.h[0] = *(const v8h*)&xt[nloc][lg * 8];
        bb.h[1] = *(const v8h*)&xt[nloc][lg * 8 + 16];

        acc = wmma_f16(a.v, bb.v, acc);
    }

    _Float16* out = outbase + (size_t)z * M * N_;
    const int n = nbase + lr;
    #pragma unroll
    for (int j = 0; j < 8; ++j) {
        const int o = obase + j + lg * 8;          // C/D layout row
        const float v = acc[j] + bias[o];
        if (transpose_out) out[(size_t)n * M + o]  = (_Float16)v;
        else               out[(size_t)o * N_ + n] = (_Float16)v;
    }
}

// ---------------------------------------------------------------------------
// Fused flash-attention: energy = Q^T K, online softmax, acc += attn * V^T.
// V tiles are double-buffered in LDS via global_load_async_to_lds_b128:
// tile t+1 is prefetched while tile t is consumed; s_wait_asynccnt 8 drains
// exactly the older batch (8 chunks/thread, in-order completion).
// Grid: (N/64, 3*B). Block: 128 (4 waves; each wave a 16-query tile with the
// full 16x256 output tile in 16 v8f accumulators).
// ---------------------------------------------------------------------------
__global__ __launch_bounds__(128)
void attn_kernel(const _Float16* __restrict__ Qt,   // [3B][N][32]
                 const _Float16* __restrict__ Kt,   // [3B][N][32]
                 const _Float16* __restrict__ Vh,   // [3B][256][N]
                 const float* __restrict__ fw,
                 float* __restrict__ out)
{
    __shared__ _Float16 vt[2][C_][40];  // double-buffered V tile [c][m]
    __shared__ _Float16 pt[4][16][40];  // per-wave attn-prob staging [n][m]

    const int lane = threadIdx.x & 31;
    const int wv   = threadIdx.x >> 5;
    const int lr   = lane & 15;
    const int lg   = lane >> 4;

    const int z = blockIdx.y;           // z = i*B + b (matches proj layout)
    const int i = z / B_, b = z % B_;

    const _Float16* Qb = Qt + (size_t)z * N_ * C8_;
    const _Float16* Kb = Kt + (size_t)z * N_ * C8_;
    const _Float16* Vb = Vh + (size_t)z * C_ * N_;

    const int nbase = blockIdx.x * 64 + wv * 16;
    const int c0 = threadIdx.x * 2;     // this thread's two V rows

    // Q A-operand (16x32, n x d), loaded once: two contiguous 8-half chunks.
    V16 qa;
    {
        const _Float16* qrow = Qb + (size_t)(nbase + lr) * C8_ + lg * 8;
        qa.h[0] = *(const v8h*)qrow;
        qa.h[1] = *(const v8h*)(qrow + 16);
    }

    v8f acc[16];
    #pragma unroll
    for (int ct = 0; ct < 16; ++ct) acc[ct] = (v8f){0,0,0,0,0,0,0,0};
    float runmax[8], runsum[8];
    #pragma unroll
    for (int j = 0; j < 8; ++j) { runmax[j] = -3.0e38f; runsum[j] = 0.f; }
    const v8f zero8 = (v8f){0,0,0,0,0,0,0,0};

    // Prologue: async-prefetch tile 0 into buffer 0 (2 rows x 4 chunks/thread)
    async_row64B(&vt[0][c0][0],     Vb + (size_t)c0 * N_);
    async_row64B(&vt[0][c0 + 1][0], Vb + (size_t)(c0 + 1) * N_);

    constexpr int ITERS = N_ / 32;
    for (int it = 0; it < ITERS; ++it) {
        const int m0  = it * 32;
        const int cur = it & 1;

        if (it + 1 < ITERS) {           // prefetch next tile into other buffer
            const _Float16* vsrc = Vb + m0 + 32;
            async_row64B(&vt[cur ^ 1][c0][0],     vsrc + (size_t)c0 * N_);
            async_row64B(&vt[cur ^ 1][c0 + 1][0], vsrc + (size_t)(c0 + 1) * N_);
            asm volatile("s_wait_asynccnt 0x8" ::: "memory");  // older 8 done
        } else {
            asm volatile("s_wait_asynccnt 0x0" ::: "memory");
        }
        __syncthreads();                // tile `it` visible to all waves

        // Energy tiles: B operand (32x16, d x m): lane col m, two 8-half d chunks (8lg, 16+8lg)
        V16 kb0, kb1;
        {
            const _Float16* k0 = Kb + (size_t)(m0 + lr) * C8_ + lg * 8;
            const _Float16* k1 = Kb + (size_t)(m0 + 16 + lr) * C8_ + lg * 8;
            kb0.h[0] = *(const v8h*)k0;  kb0.h[1] = *(const v8h*)(k0 + 16);
            kb1.h[0] = *(const v8h*)k1;  kb1.h[1] = *(const v8h*)(k1 + 16);
        }
        v8f e0 = wmma_f16(qa.v, kb0.v, zero8);
        v8f e1 = wmma_f16(qa.v, kb1.v, zero8);

        // Online softmax: row r lives at (vgpr j, lane group lg); the 16 m
        // columns are lanes within the group -> xor masks 1,2,4,8 suffice.
        float scale[8];
        #pragma unroll
        for (int j = 0; j < 8; ++j) {
            float t = fmaxf(e0[j], e1[j]);
            t = fmaxf(t, __shfl_xor(t, 1, 32));
            t = fmaxf(t, __shfl_xor(t, 2, 32));
            t = fmaxf(t, __shfl_xor(t, 4, 32));
            t = fmaxf(t, __shfl_xor(t, 8, 32));
            const float nm = fmaxf(runmax[j], t);
            scale[j]  = __expf(runmax[j] - nm);
            runmax[j] = nm;
            const float p0 = __expf(e0[j] - nm);
            const float p1 = __expf(e1[j] - nm);
            float s = p0 + p1;
            s += __shfl_xor(s, 1, 32);
            s += __shfl_xor(s, 2, 32);
            s += __shfl_xor(s, 4, 32);
            s += __shfl_xor(s, 8, 32);
            runsum[j] = runsum[j] * scale[j] + s;
            e0[j] = p0; e1[j] = p1;
        }
        #pragma unroll
        for (int ct = 0; ct < 16; ++ct)
            #pragma unroll
            for (int j = 0; j < 8; ++j)
                acc[ct][j] *= scale[j];

        // C/D layout -> A-operand layout via wave-private LDS region.
        // DS ops are in-order within a wave; a scheduling barrier suffices.
        #pragma unroll
        for (int j = 0; j < 8; ++j) {
            pt[wv][j + lg * 8][lr]      = (_Float16)e0[j];
            pt[wv][j + lg * 8][16 + lr] = (_Float16)e1[j];
        }
        __builtin_amdgcn_wave_barrier();

        V16 pa;
        pa.h[0] = *(const v8h*)&pt[wv][lr][lg * 8];
        pa.h[1] = *(const v8h*)&pt[wv][lr][16 + lg * 8];

        // acc[n, c] += P(16x32) x V^T(32x16) across 16 c-tiles
        #pragma unroll
        for (int ct = 0; ct < 16; ++ct) {
            V16 vb;
            const int c = ct * 16 + lr;
            vb.h[0] = *(const v8h*)&vt[cur][c][lg * 8];
            vb.h[1] = *(const v8h*)&vt[cur][c][lg * 8 + 16];
            acc[ct] = wmma_f16(pa.v, vb.v, acc[ct]);
        }

        __syncthreads();   // all reads of vt[cur] done before it is re-staged
    }

    // Normalize, weight by fusion_weights[i]/3, accumulate across streams.
    const float wgt = fw[i] * (1.0f / 3.0f);
    #pragma unroll
    for (int ct = 0; ct < 16; ++ct) {
        const int c = ct * 16 + lr;
        #pragma unroll
        for (int j = 0; j < 8; ++j) {
            const int n = nbase + j + lg * 8;
            const float v = acc[ct][j] * wgt / runsum[j];
            atomicAdd(&out[((size_t)b * C_ + c) * N_ + n], v);
        }
    }
}

extern "C" void kernel_launch(void* const* d_in, const int* in_sizes, int n_in,
                              void* d_out, int out_size, void* d_ws, size_t ws_size,
                              hipStream_t stream) {
    (void)in_sizes; (void)n_in; (void)ws_size;
    const float* s0 = (const float*)d_in[0];
    const float* s1 = (const float*)d_in[1];
    const float* s2 = (const float*)d_in[2];
    const float* Wq = (const float*)d_in[3];
    const float* bq = (const float*)d_in[4];
    const float* Wk = (const float*)d_in[5];
    const float* bk = (const float*)d_in[6];
    const float* Wv = (const float*)d_in[7];
    const float* bv = (const float*)d_in[8];
    const float* fw = (const float*)d_in[9];
    float* out = (float*)d_out;

    char* ws = (char*)d_ws;
    const size_t qBytes = (size_t)3 * B_ * N_ * C8_ * sizeof(_Float16); // 4.9 MB
    _Float16* Qt = (_Float16*)(ws);
    _Float16* Kt = (_Float16*)(ws + qBytes);
    _Float16* Vh = (_Float16*)(ws + 2 * qBytes);                        // 39.3 MB

    hipMemsetAsync(d_out, 0, (size_t)out_size * sizeof(float), stream);

    dim3 blk(128);
    // Q from stream i, K/V from stream (i+1)%3; Q,K stored [n][32], V [256][n]
    proj_kernel<<<dim3(N_ / 64, C8_ / 16, 3 * B_), blk, 0, stream>>>(
        s0, s1, s2, Wq, bq, Qt, C8_, /*shift=*/0, /*transpose=*/1);
    proj_kernel<<<dim3(N_ / 64, C8_ / 16, 3 * B_), blk, 0, stream>>>(
        s0, s1, s2, Wk, bk, Kt, C8_, /*shift=*/1, /*transpose=*/1);
    proj_kernel<<<dim3(N_ / 64, C_ / 16, 3 * B_), blk, 0, stream>>>(
        s0, s1, s2, Wv, bv, Vh, C_,  /*shift=*/1, /*transpose=*/0);

    attn_kernel<<<dim3(N_ / 64, 3 * B_), blk, 0, stream>>>(Qt, Kt, Vh, fw, out);
}
